// Seq2SeqLSTM_36550171689022
// MI455X (gfx1250) — matrix operand order repaired and checked by reading, not compiled, not run on hardware
//
#include <hip/hip_runtime.h>

// ---------------------------------------------------------------------------
// Seq2Seq LSTM (B=2048, S=288, H=512, T=144) for MI455X / gfx1250.
// Batch-parallel persistent blocks (16 rows each), bf16 WMMA
// (v_wmma_f32_16x16x32_bf16) for all h@W.T GEMMs, fp32 gate math, c-state in
// VGPRs, h-state in LDS, weights pre-swizzled into the exact per-lane WMMA
// B-matrix layout so every B tile is one contiguous 32B load (L2 resident).
// Round 2: 16 waves/block, 8 N-tiles per wave -> <256 VGPRs (no vgpr-msb
// switching in the inner loop), 4 waves/SIMD to hide L2 load latency.
// ---------------------------------------------------------------------------

typedef __attribute__((ext_vector_type(8)))  float  v8f;
typedef __attribute__((ext_vector_type(8)))  __bf16 v8bf;
typedef __attribute__((ext_vector_type(16))) __bf16 v16bf;

#define HH   512          // hidden size
#define SS   288          // encoder steps
#define TT   144          // decoder steps
#define BB   2048         // batch
#define BT   16           // batch rows per block
#define NTH  512          // threads per block (16 wave32 waves)
#define WELEMS (2048*512) // elements per weight matrix

__device__ __forceinline__ unsigned short f2bf(float f) {
    unsigned int u = __float_as_uint(f);
    u += 0x7FFFu + ((u >> 16) & 1u);      // round-to-nearest-even
    return (unsigned short)(u >> 16);
}
__device__ __forceinline__ float bf2f(unsigned short h) {
    return __uint_as_float(((unsigned int)h) << 16);
}
__device__ __forceinline__ float sigf(float x) {
    return 1.0f / (1.0f + __expf(-x));
}
__device__ __forceinline__ float tanhfast(float x) {
    x = fminf(fmaxf(x, -15.0f), 15.0f);
    float e = __expf(2.0f * x);
    return (e - 1.0f) / (e + 1.0f);
}

// ---------------------------------------------------------------------------
// Prep kernel 1: convert a (2048x512) fp32 weight into bf16, pre-swizzled into
// a per-lane WMMA B order with contiguous k per lane half (16-bit B 32x16 tiles):
//   tile (nt, kc): lane l, elem j  ->  W[n][k],
//   n = nt*16 + (l&15),  k = kc*32 + (l>=16 ? 16 : 0) + j
// Memory order: addr = mat*WELEMS + (nt*16+kc)*512 + lane*16 + j
// ---------------------------------------------------------------------------
__global__ void prep_weights(const float* __restrict__ w0, const float* __restrict__ w1,
                             const float* __restrict__ w2, const float* __restrict__ w3,
                             const float* __restrict__ w4, const float* __restrict__ w5,
                             unsigned short* __restrict__ wt_all) {
    const int idx  = blockIdx.x * 256 + threadIdx.x;    // 6 * 2^20 total
    const int mat  = idx >> 20;
    const int rem  = idx & (WELEMS - 1);
    const int tile = rem >> 9;                          // /512
    const int lane = (rem >> 4) & 31;
    const int j    = rem & 15;
    const int nt   = tile >> 4, kc = tile & 15;
    const int n    = nt * 16 + (lane & 15);
    const int k    = kc * 32 + ((lane >> 4) << 4) + j;
    const float* src = (mat == 0) ? w0 : (mat == 1) ? w1 : (mat == 2) ? w2
                     : (mat == 3) ? w3 : (mat == 4) ? w4 : w5;
    wt_all[idx] = f2bf(src[n * HH + k]);
}

// Prep kernel 2: combined biases be = bih + bhh, four vectors of 2048.
__global__ void prep_bias(const float* __restrict__ ei0, const float* __restrict__ eh0,
                          const float* __restrict__ ei1, const float* __restrict__ eh1,
                          const float* __restrict__ di0, const float* __restrict__ dh0,
                          const float* __restrict__ di1, const float* __restrict__ dh1,
                          float* __restrict__ be_all) {
    const int idx = blockIdx.x * 256 + threadIdx.x;     // 8192 total
    const int w = idx >> 11, j = idx & 2047;
    float v;
    switch (w) {
        case 0:  v = ei0[j] + eh0[j]; break;
        case 1:  v = ei1[j] + eh1[j]; break;
        case 2:  v = di0[j] + dh0[j]; break;
        default: v = di1[j] + dh1[j]; break;
    }
    be_all[idx] = v;
}

// ---------------------------------------------------------------------------
// GEMM: acc[8] += hA(16x512 bf16, LDS) @ wt(512x2048 pre-swizzled bf16).
// Wave w (0..15) owns N tiles {gi*512 + w*32 + k*16 : gi in 0..3, k in 0..1}.
// A per lane (contiguous k, same order as B): lanes 0-15 hold K {0..15},
// lanes 16-31 hold K {16..31} of each 32-wide chunk -> two 16B ds loads.
// ---------------------------------------------------------------------------
__device__ __forceinline__ void gemm_accum(const unsigned short* __restrict__ wt,
                                           const unsigned short* hA,  // LDS [16][512]
                                           int wave, int lane, v8f acc[8]) {
    const int m    = lane & 15;
    const int offA = (lane & 16) ? 16 : 0;
    const unsigned short* aRow = hA + m * HH;
    for (int kc = 0; kc < 16; ++kc) {
        const int kb = kc << 5;
        const v8bf a0 = *(const v8bf*)(aRow + kb + offA);
        const v8bf a1 = *(const v8bf*)(aRow + kb + 8 + offA);
        v16bf A;
        #pragma unroll
        for (int i = 0; i < 8; ++i) { A[i] = a0[i]; A[i + 8] = a1[i]; }
        #pragma unroll
        for (int s = 0; s < 8; ++s) {
            const int gi = s >> 1, k = s & 1;
            const int nt = gi * 32 + wave * 2 + k;
            const v16bf Bv = *(const v16bf*)(wt + (nt * 16 + kc) * 512 + lane * 16);
            acc[s] = __builtin_amdgcn_wmma_f32_16x16x32_bf16(
                false, A, false, Bv, (short)0, acc[s], false, false);
        }
    }
}

// ---------------------------------------------------------------------------
// LSTM cell elementwise. C-matrix layout: VGPR v, lane l -> row v+8*(l>=16),
// col (l&15) within each 16-wide tile. c-state lives in VGPRs (16 per layer).
// ---------------------------------------------------------------------------
template <bool HAS_X>
__device__ __forceinline__ void lstm_elem(v8f acc[8], float c[16],
                                          const float* __restrict__ be,
                                          const float* __restrict__ wih,
                                          const float xr[8],
                                          unsigned short* hOut,  // LDS [16][512]
                                          int wave, int lane) {
    const int col0 = wave * 32 + (lane & 15);
    const int rh   = (lane >> 4) * 8;
    #pragma unroll
    for (int k = 0; k < 2; ++k) {
        const int j = col0 + k * 16;
        const float bi = be[j], bff = be[HH + j], bg = be[2 * HH + j], bo = be[3 * HH + j];
        float wi = 0.f, wf = 0.f, wg = 0.f, wo = 0.f;
        if (HAS_X) { wi = wih[j]; wf = wih[HH + j]; wg = wih[2 * HH + j]; wo = wih[3 * HH + j]; }
        #pragma unroll
        for (int v = 0; v < 8; ++v) {
            float xi = acc[k][v]     + bi;      // gate i tiles: s = 0..1
            float xf = acc[2 + k][v] + bff;     // gate f tiles: s = 2..3
            float xg = acc[4 + k][v] + bg;      // gate g tiles: s = 4..5
            float xo = acc[6 + k][v] + bo;      // gate o tiles: s = 6..7
            if (HAS_X) {
                const float xx = xr[v];
                xi += xx * wi; xf += xx * wf; xg += xx * wg; xo += xx * wo;
            }
            const float ii = sigf(xi), ff = sigf(xf);
            const float gg = tanhfast(xg), oo = sigf(xo);
            const float cc = ff * c[k * 8 + v] + ii * gg;
            c[k * 8 + v] = cc;
            hOut[(rh + v) * HH + j] = f2bf(oo * tanhfast(cc));
        }
    }
}

// One full 2-layer LSTM time step for this block's 16 batch rows.
__device__ __forceinline__ void lstm_step(
    const unsigned short* __restrict__ wt_hh0,
    const unsigned short* __restrict__ wt_ih1,
    const unsigned short* __restrict__ wt_hh1,
    const float* __restrict__ be0, const float* __restrict__ be1,
    const float* __restrict__ wih0,
    unsigned short* hA0, unsigned short* hA1, const float* xvs,
    float c0[16], float c1[16], int wave, int lane) {
    v8f acc[8];
    const v8f z = {};
    #pragma unroll
    for (int s = 0; s < 8; ++s) acc[s] = z;
    gemm_accum(wt_hh0, hA0, wave, lane, acc);     // reads hA0 (old h0)
    __syncthreads();

    float xr[8];
    const int rh = (lane >> 4) * 8;
    #pragma unroll
    for (int v = 0; v < 8; ++v) xr[v] = xvs[rh + v];
    lstm_elem<true>(acc, c0, be0, wih0, xr, hA0, wave, lane);   // writes new h0
    __syncthreads();

    #pragma unroll
    for (int s = 0; s < 8; ++s) acc[s] = z;
    gemm_accum(wt_ih1, hA0, wave, lane, acc);     // h0_new @ Wih1.T
    gemm_accum(wt_hh1, hA1, wave, lane, acc);     // h1_old @ Whh1.T
    __syncthreads();
    lstm_elem<false>(acc, c1, be1, nullptr, nullptr, hA1, wave, lane);  // writes new h1
    __syncthreads();
}

// ---------------------------------------------------------------------------
// Main persistent kernel: 128 blocks x 512 threads (16 wave32 waves).
// ---------------------------------------------------------------------------
__global__ __launch_bounds__(NTH) void seq2seq_main(
    const float* __restrict__ x,
    const unsigned short* __restrict__ wt_all,   // 6 swizzled bf16 matrices
    const float* __restrict__ be_all,            // be0,be1,bd0,bd1
    const float* __restrict__ enc_wih0,          // (2048,)
    const float* __restrict__ dec_wih0,          // (2048,)
    const float* __restrict__ fcW,               // (512,)
    const float* __restrict__ fcBp,              // (1,)
    float* __restrict__ out) {                   // (B, T)
    __shared__ unsigned short hA0[BT * HH];      // 16 KB, bf16 h0
    __shared__ unsigned short hA1[BT * HH];      // 16 KB, bf16 h1
    __shared__ float xvs[BT];                    // scalar input per row
    __shared__ float fcred[NTH];                 // fc partial sums

    const int tid  = threadIdx.x;
    const int wave = tid >> 5, lane = tid & 31;
    const int rowBase = blockIdx.x * BT;

    for (int i = tid; i < BT * HH; i += NTH) { hA0[i] = 0; hA1[i] = 0; }
    float c0[16], c1[16];
    #pragma unroll
    for (int i = 0; i < 16; ++i) { c0[i] = 0.f; c1[i] = 0.f; }
    __syncthreads();

    const unsigned short* wt_e_hh0 = wt_all + 0 * WELEMS;
    const unsigned short* wt_e_ih1 = wt_all + 1 * WELEMS;
    const unsigned short* wt_e_hh1 = wt_all + 2 * WELEMS;
    const unsigned short* wt_d_hh0 = wt_all + 3 * WELEMS;
    const unsigned short* wt_d_ih1 = wt_all + 4 * WELEMS;
    const unsigned short* wt_d_hh1 = wt_all + 5 * WELEMS;
    const float* be0 = be_all;
    const float* be1 = be_all + 2048;
    const float* bd0 = be_all + 4096;
    const float* bd1 = be_all + 6144;
    const float fcb = *fcBp;

    // -------- encoder --------
    for (int t = 0; t < SS; ++t) {
        if (tid < BT) xvs[tid] = x[(size_t)(rowBase + tid) * SS + t];  // visible by 1st barrier
        lstm_step(wt_e_hh0, wt_e_ih1, wt_e_hh1, be0, be1, enc_wih0,
                  hA0, hA1, xvs, c0, c1, wave, lane);
    }
    // xvs now holds x[:, S-1] == dec_in

    // -------- decoder --------
    for (int t = 0; t < TT; ++t) {
        lstm_step(wt_d_hh0, wt_d_ih1, wt_d_hh1, bd0, bd1, dec_wih0,
                  hA0, hA1, xvs, c0, c1, wave, lane);
        // fc: pred = h1 @ fc_W.T + fc_b ; feed back as next input
        {
            const int r = tid >> 5, seg = tid & 31;   // 32 partials per row
            float p = 0.f;
            for (int q = 0; q < 16; ++q) {
                const int j = seg * 16 + q;
                p += bf2f(hA1[r * HH + j]) * fcW[j];
            }
            fcred[r * 32 + seg] = p;
            __syncthreads();
            if (tid < BT) {
                float s = fcb;
                #pragma unroll
                for (int q = 0; q < 32; ++q) s += fcred[tid * 32 + q];
                xvs[tid] = s;
                out[(size_t)(rowBase + tid) * TT + t] = s;
            }
            __syncthreads();
        }
    }
}

extern "C" void kernel_launch(void* const* d_in, const int* in_sizes, int n_in,
                              void* d_out, int out_size, void* d_ws, size_t ws_size,
                              hipStream_t stream) {
    (void)in_sizes; (void)n_in; (void)out_size; (void)ws_size;
    const float* x        = (const float*)d_in[0];
    const float* e_wih0   = (const float*)d_in[1];
    const float* e_whh0   = (const float*)d_in[2];
    const float* e_bih0   = (const float*)d_in[3];
    const float* e_bhh0   = (const float*)d_in[4];
    const float* e_wih1   = (const float*)d_in[5];
    const float* e_whh1   = (const float*)d_in[6];
    const float* e_bih1   = (const float*)d_in[7];
    const float* e_bhh1   = (const float*)d_in[8];
    const float* d_wih0   = (const float*)d_in[9];
    const float* d_whh0   = (const float*)d_in[10];
    const float* d_bih0   = (const float*)d_in[11];
    const float* d_bhh0   = (const float*)d_in[12];
    const float* d_wih1   = (const float*)d_in[13];
    const float* d_whh1   = (const float*)d_in[14];
    const float* d_bih1   = (const float*)d_in[15];
    const float* d_bhh1   = (const float*)d_in[16];
    const float* fcW      = (const float*)d_in[17];
    const float* fcB      = (const float*)d_in[18];
    float* out            = (float*)d_out;

    unsigned short* wt_all = (unsigned short*)d_ws;                     // 12 MB bf16
    float* be_all = (float*)((char*)d_ws + (size_t)6 * WELEMS * 2);     // 32 KB fp32

    // Swizzle weights: order must match seq2seq_main's slot assignment.
    prep_weights<<<(6 * WELEMS) / 256, 256, 0, stream>>>(
        e_whh0, e_wih1, e_whh1, d_whh0, d_wih1, d_whh1, wt_all);
    prep_bias<<<(4 * 2048) / 256, 256, 0, stream>>>(
        e_bih0, e_bhh0, e_bih1, e_bhh1, d_bih0, d_bhh0, d_bih1, d_bhh1, be_all);

    seq2seq_main<<<BB / BT, NTH, 0, stream>>>(
        x, wt_all, be_all, e_wih0, d_wih0, fcW, fcB, out);
}
